// RWN_16329465659692
// MI455X (gfx1250) — matrix operand order repaired and checked by reading, compiled, not run on hardware
//
#include <hip/hip_runtime.h>
#include <hip/hip_bf16.h>

// ---------------------------------------------------------------------------
// MI455X (gfx1250) implementation.
//
// Pass 1 (bandwidth-bound, 275MB of reads): f[b,i,j] = sum_c w[c]*feats[c]
// streamed with float4 loads, scalar-broadcast weights.
//
// Pass 2: 7x7 neighborhood softmax attention. One wave32 per
// (batch, row, 16-column block). The 49-weight aggregation over 21 channels
// is cast as D(16x16) = A(16x160) * X(160x16) with A the banded softmax
// weight matrix over the 7x22 window and X the staged x2 tile, executed as
// 5 chained v_wmma_f32_16x16x32_f16 per channel tile (2 tiles for 21 ch).
// LDS operands are laid out K-contiguous per lane so fragment assembly is
// ds_load_b128 (16B per lane, conflict-free 336B row stride), not u16 gather.
// x2 (44MB) fits in the 192MB L2, so the 7x7 halo re-reads stay on-chip.
// ---------------------------------------------------------------------------

typedef __attribute__((ext_vector_type(16))) _Float16 v16h;
typedef __attribute__((ext_vector_type(8)))  _Float16 v8h;
typedef __attribute__((ext_vector_type(2)))  _Float16 h2;
typedef __attribute__((ext_vector_type(8)))  float    v8f;

#define N_      256
#define NN      (N_ * N_)
#define R_      3
#define B_      8
#define NCLS    21
#define W7      7              // 2R+1
#define TJ      16             // pixels (columns) per wave tile
#define WIN     22             // TJ + 2R : column window
#define KH      (W7 * WIN)     // 154 : logical K extent
#define KPAD    160            // K padded to 5 x 32
#define KSTRIDE 168            // LDS K stride in halves: 336B = 21*16B (odd*16B)
#define CROWS   32             // channel rows in Xs (21 real + zero pad)

// ---------------- Pass 1: 131-channel weighted reduction -------------------
__global__ __launch_bounds__(256)
void f_reduce_kernel(const float* __restrict__ x,
                     const float* __restrict__ c1,
                     const float* __restrict__ c2,
                     const float* __restrict__ w,
                     float* __restrict__ f) {
    int idx = blockIdx.x * blockDim.x + threadIdx.x;     // over B*NN/4
    int b   = idx / (NN / 4);
    int q   = idx % (NN / 4);                            // float4 index in plane

    const float4* xb  = (const float4*)(x  + (size_t)b * 3  * NN);
    const float4* c1b = (const float4*)(c1 + (size_t)b * 64 * NN);
    const float4* c2b = (const float4*)(c2 + (size_t)b * 64 * NN);

    float4 acc = make_float4(0.f, 0.f, 0.f, 0.f);
#pragma unroll
    for (int c = 0; c < 3; c++) {
        float wc = w[c];
        float4 v = xb[(size_t)c * (NN / 4) + q];
        acc.x += wc * v.x; acc.y += wc * v.y; acc.z += wc * v.z; acc.w += wc * v.w;
    }
#pragma unroll 8
    for (int c = 0; c < 64; c++) {
        float wc = w[3 + c];
        float4 v = c1b[(size_t)c * (NN / 4) + q];
        acc.x += wc * v.x; acc.y += wc * v.y; acc.z += wc * v.z; acc.w += wc * v.w;
    }
#pragma unroll 8
    for (int c = 0; c < 64; c++) {
        float wc = w[67 + c];
        float4 v = c2b[(size_t)c * (NN / 4) + q];
        acc.x += wc * v.x; acc.y += wc * v.y; acc.z += wc * v.z; acc.w += wc * v.w;
    }
    ((float4*)f)[(size_t)b * (NN / 4) + q] = acc;
}

// ---------------- Pass 2: neighborhood softmax + WMMA aggregation ----------
__global__ __launch_bounds__(32)
void attn_wmma_kernel(const float* __restrict__ f,
                      const float* __restrict__ x2,
                      float* __restrict__ out) {
    __shared__ float fs[W7][WIN];                            // f halo tile
    __shared__ __align__(16) _Float16 As[TJ][KSTRIDE];       // weights, [pixel][k]
    __shared__ __align__(16) _Float16 Xs[CROWS][KSTRIDE];    // x2 tile,  [chan][k]

    const int lane = threadIdx.x;
    const int wg   = blockIdx.x;                 // B * N * (N/TJ) = 32768
    const int jblk = wg & 15;                    // N/TJ = 16
    const int i    = (wg >> 4) & (N_ - 1);
    const int b    = wg >> 12;
    const int j0   = jblk * TJ;

    // Zero As (banded gaps) and Xs (K pad + channel pad rows) with b128 stores.
    {
        const uint4 z = make_uint4(0u, 0u, 0u, 0u);
        uint4* pa = (uint4*)&As[0][0];                       // 16*168*2/16 = 336
        for (int t = lane; t < (TJ * KSTRIDE) / 8; t += 32) pa[t] = z;
        uint4* px = (uint4*)&Xs[0][0];                       // 32*168*2/16 = 672
        for (int t = lane; t < (CROWS * KSTRIDE) / 8; t += 32) px[t] = z;
    }

    // Stage f halo: rows i-3..i+3, cols j0-3..j0+18 (0 outside; masked later).
    const float* fb = f + (size_t)b * NN;
    for (int e = lane; e < KH; e += 32) {
        int di = e / WIN, q = e - di * WIN;
        int gi = i - R_ + di, gj = j0 - R_ + q;
        float v = 0.f;
        if (gi >= 0 && gi < N_ && gj >= 0 && gj < N_) v = fb[gi * N_ + gj];
        fs[di][q] = v;
    }
    __syncthreads();

    // Stage x2 tile as f16: Xs[c][di*22+q] = x2[b,c,i-3+di,j0-3+q] (0 OOB).
    // Two adjacent columns per lane -> one packed ds_store_b32; consecutive
    // lanes touch consecutive gj -> coalesced global reads.
    const float* x2b = x2 + (size_t)b * NCLS * NN;
    for (int t = lane; t < (KH / 2) * NCLS; t += 32) {
        int c  = t / (KH / 2), k = (t % (KH / 2)) * 2;
        int di = k / WIN, q = k - di * WIN;                  // q even, q+1 <= 21
        int gi = i - R_ + di, gj = j0 - R_ + q;
        float v0 = 0.f, v1 = 0.f;
        if (gi >= 0 && gi < N_) {
            const float* row = x2b + (size_t)c * NN + (size_t)gi * N_;
            if (gj >= 0 && gj < N_)         v0 = row[gj];
            if (gj + 1 >= 0 && gj + 1 < N_) v1 = row[gj + 1];
        }
        h2 hv = { (_Float16)v0, (_Float16)v1 };
        *(h2*)&Xs[c][k] = hv;                                // 4B-aligned (k even)
    }
    __syncthreads();

    // Per-pixel 49-way softmax on lanes 0..15 (cheap vs. memory traffic).
    if (lane < TJ) {
        const int   p  = lane;
        const float fc = fs[R_][p + R_];
        float sc[W7 * W7];
        float m = -3.4e38f;
#pragma unroll
        for (int di = 0; di < W7; di++) {
            int  gi    = i - R_ + di;
            bool rowok = (gi >= 0) && (gi < N_);
#pragma unroll
            for (int dj = 0; dj < W7; dj++) {
                int  gj = j0 + p - R_ + dj;
                bool ok = rowok && (gj >= 0) && (gj < N_);
                float s = ok ? fabsf(fc - fs[di][p + dj]) : -3.4e38f;
                sc[di * W7 + dj] = s;
                if (ok) m = fmaxf(m, s);
            }
        }
        float sum = 0.f;
#pragma unroll
        for (int k = 0; k < W7 * W7; k++) {
            float e = __expf(sc[k] - m);      // -3.4e38 - m underflows to 0
            sum += e;
            sc[k] = e;
        }
        const float inv = 1.f / sum;          // center is always valid: sum>0
#pragma unroll
        for (int di = 0; di < W7; di++)
#pragma unroll
            for (int dj = 0; dj < W7; dj++)
                As[p][di * WIN + p + dj] = (_Float16)(sc[di * W7 + dj] * inv);
    }
    __syncthreads();

    // WMMA: D(16x16) += A(16x32) * X(32x16), 5 K-chunks, 2 channel tiles.
    // Fragment layouts per CDNA5 ISA 7.12.2 (wave32):
    //   A (f16 16x32): lanes 0-15 = rows; h[0..7] -> K=0..7 (+8 upper half-wave),
    //                  h[8..15] -> K=16..23 (+8 upper half-wave)
    //   B (f16 32x16): lane%16 = column, K order as A; code uses h[t] -> K = t (+16 upper) for both
    //   C/D (f32 16x16): VGPR r -> row r (+8 upper half-wave), lane%16 = column
    // K-contiguous LDS rows make each operand two 16B ds_load_b128 per lane.
    const int hsel = lane >> 4;
    const int l16  = lane & 15;
    v8f acc0 = {};
    v8f acc1 = {};
#pragma unroll
    for (int ch = 0; ch < KPAD / 32; ch++) {
        const int kA = ch * 32 + hsel * 16;   // A: K run 0..15 / 16..31
        const int kB = ch * 32 + hsel * 16;   // B: K run 0..15 / 16..31
        v8h alo = *(const v8h*)&As[l16][kA];
        v8h ahi = *(const v8h*)&As[l16][kA + 8];
        v8h b0l = *(const v8h*)&Xs[l16][kB];
        v8h b0h = *(const v8h*)&Xs[l16][kB + 8];
        v8h b1l = *(const v8h*)&Xs[16 + l16][kB];
        v8h b1h = *(const v8h*)&Xs[16 + l16][kB + 8];
        v16h a  = __builtin_shufflevector(alo, ahi, 0,1,2,3,4,5,6,7,8,9,10,11,12,13,14,15);
        v16h b0 = __builtin_shufflevector(b0l, b0h, 0,1,2,3,4,5,6,7,8,9,10,11,12,13,14,15);
        v16h b1 = __builtin_shufflevector(b1l, b1h, 0,1,2,3,4,5,6,7,8,9,10,11,12,13,14,15);
        acc0 = __builtin_amdgcn_wmma_f32_16x16x32_f16(false, a, false, b0,
                                                      (short)0, acc0, false, false);
        acc1 = __builtin_amdgcn_wmma_f32_16x16x32_f16(false, a, false, b1,
                                                      (short)0, acc1, false, false);
    }

    // Scatter D: out[b, i*N + j0 + p, c], p = row of D, c = column (+16 tile 1).
    float* ob = out + ((size_t)b * NN + (size_t)i * N_ + j0) * NCLS;
#pragma unroll
    for (int r = 0; r < 8; r++) {
        int p = r + hsel * 8;
        ob[(size_t)p * NCLS + l16] = acc0[r];
        if (16 + l16 < NCLS)
            ob[(size_t)p * NCLS + 16 + l16] = acc1[r];
    }
}

// ---------------------------------------------------------------------------
extern "C" void kernel_launch(void* const* d_in, const int* in_sizes, int n_in,
                              void* d_out, int out_size, void* d_ws, size_t ws_size,
                              hipStream_t stream) {
    (void)in_sizes; (void)n_in; (void)out_size; (void)ws_size;
    const float* x     = (const float*)d_in[0];   // (B, 3, N, N)
    const float* conv1 = (const float*)d_in[1];   // (B, 64, N, N)
    const float* conv2 = (const float*)d_in[2];   // (B, 64, N, N)
    const float* x2    = (const float*)d_in[3];   // (B, NCLS, N, N)
    const float* wconv = (const float*)d_in[4];   // (131,)
    float*       out   = (float*)d_out;           // (B, N*N, NCLS)
    float*       fbuf  = (float*)d_ws;            // B*N*N floats = 2MB scratch

    // Pass 1: 131-channel reduction, float4 streaming.
    {
        int total = (B_ * NN) / 4;
        f_reduce_kernel<<<total / 256, 256, 0, stream>>>(x, conv1, conv2, wconv, fbuf);
    }
    // Pass 2: one wave32 per (b, row, 16-col block).
    {
        int blocks = B_ * N_ * (N_ / TJ);         // 32768
        attn_wmma_kernel<<<blocks, 32, 0, stream>>>(fbuf, x2, out);
    }
}
